// SparseConnectedLayer_19868518711680
// MI455X (gfx1250) — compile-verified, matrix operand order checked by reading, not run on hardware
//
#include <hip/hip_runtime.h>

// ---------------------------------------------------------------------------
// SparseConnectedLayer: out[b][o] = sum_k x[b][k] * (w[o][k]*mask[o][k]) + bias[o]
// M=8192, N=4096, K=4096, fp32 in/out.
//
// Pre-pass folds mask and converts to bf16 (operands then fit in the 192MB L2).
// GEMM: v_wmma_f32_16x16x32_bf16, 64x64 wave tiles (4x4 accumulators),
// double-buffered LDS filled by CDNA5 async global->LDS copies (ASYNCcnt).
// ---------------------------------------------------------------------------

typedef __attribute__((ext_vector_type(16))) __bf16 v16bf;
typedef __attribute__((ext_vector_type(8)))  float  v8f;

static constexpr int Mdim = 8192;
static constexpr int Ndim = 4096;
static constexpr int Kdim = 4096;

__device__ __forceinline__ unsigned short f2bf(float f) {
    unsigned u = __builtin_bit_cast(unsigned, f);
    u += 0x7fffu + ((u >> 16) & 1u);
    return (unsigned short)(u >> 16);
}
__device__ __forceinline__ unsigned pack2(float lo, float hi) {
    return (unsigned)f2bf(lo) | ((unsigned)f2bf(hi) << 16);
}

// ---- pre-pass: x (fp32) -> bf16, 8 elements/thread ------------------------
__global__ __launch_bounds__(256) void pack_x_bf16(const float* __restrict__ x,
                                                   unsigned short* __restrict__ xb) {
    size_t gid = (size_t)blockIdx.x * 256 + threadIdx.x;
    const float4* x4 = (const float4*)x;
    float4 a = x4[gid * 2 + 0];
    float4 b = x4[gid * 2 + 1];
    uint4 q;
    q.x = pack2(a.x, a.y);
    q.y = pack2(a.z, a.w);
    q.z = pack2(b.x, b.y);
    q.w = pack2(b.z, b.w);
    ((uint4*)xb)[gid] = q;
}

// ---- pre-pass: w*mask (fp32) -> bf16, 8 elements/thread -------------------
__global__ __launch_bounds__(256) void pack_w_bf16(const float* __restrict__ w,
                                                   const float* __restrict__ m,
                                                   unsigned short* __restrict__ wb) {
    size_t gid = (size_t)blockIdx.x * 256 + threadIdx.x;
    const float4* w4 = (const float4*)w;
    const float4* m4 = (const float4*)m;
    float4 a = w4[gid * 2 + 0], am = m4[gid * 2 + 0];
    float4 b = w4[gid * 2 + 1], bm = m4[gid * 2 + 1];
    uint4 q;
    q.x = pack2(a.x * am.x, a.y * am.y);
    q.y = pack2(a.z * am.z, a.w * am.w);
    q.z = pack2(b.x * bm.x, b.y * bm.y);
    q.w = pack2(b.z * bm.z, b.w * bm.w);
    ((uint4*)wb)[gid] = q;
}

// ---- main GEMM: C[M,N] = A[M,K] * B[N,K]^T + bias -------------------------
// Block tile 256(M) x 128(N), BK=32, 256 threads = 8 wave32 (4 in M x 2 in N).
// Each wave: 64x64 tile = 4x4 WMMA accumulators (frag:wmma = 0.5).
__global__ __launch_bounds__(256) void sparse_gemm_wmma(
    const unsigned short* __restrict__ A,   // bf16 [M,K] row-major
    const unsigned short* __restrict__ B,   // bf16 [N,K] row-major
    const float* __restrict__ bias,         // [N]
    float* __restrict__ C)                  // fp32 [M,N] row-major
{
    constexpr int BM = 256, BN = 128, BK = 32;
    constexpr int LDK  = 40;                // padded row stride (bf16): 20 dwords,
                                            // 16-lane b128 reads -> 16 distinct banks
    constexpr int ABUF = BM * LDK;          // 10240 ushort = 20 KB per buffer
    constexpr int BBUF = BN * LDK;          // 5120 ushort  = 10 KB per buffer

    __shared__ alignas(16) unsigned short sA[2 * ABUF];   // 40 KB
    __shared__ alignas(16) unsigned short sB[2 * BBUF];   // 20 KB

    const int tid  = threadIdx.x;
    const int lane = tid & 31;
    const int wave = tid >> 5;
    const int wm   = wave >> 1;   // 0..3 : wave row (64 M rows)
    const int wn   = wave & 1;    // 0..1 : wave col (64 N cols)
    const int lr   = lane & 15;
    const int lh   = lane >> 4;

    const int m0 = blockIdx.y * BM;
    const int n0 = blockIdx.x * BN;

    // ---- per-thread async-copy slots ----
    // A: 256 rows x 4 chunks = 1024 chunks -> 4/thread; B: 512 chunks -> 2/thread
    const unsigned short* gA[4]; unsigned lA[4];
    const unsigned short* gB[2]; unsigned lB[2];
    #pragma unroll
    for (int j = 0; j < 4; ++j) {
        int lin = tid + j * 256;
        int row = lin >> 2, cc = (lin & 3) * 8;
        gA[j] = A + (size_t)(m0 + row) * Kdim + cc;
        lA[j] = (unsigned)(uintptr_t)(sA + row * LDK + cc);
    }
    #pragma unroll
    for (int j = 0; j < 2; ++j) {
        int lin = tid + j * 256;
        int row = lin >> 2, cc = (lin & 3) * 8;
        gB[j] = B + (size_t)(n0 + row) * Kdim + cc;
        lB[j] = (unsigned)(uintptr_t)(sB + row * LDK + cc);
    }
    constexpr unsigned ABUF_BYTES = ABUF * 2;   // 20480
    constexpr unsigned BBUF_BYTES = BBUF * 2;   // 10240

    auto issue = [&](int kb, int buf) {
        const unsigned oA = buf ? ABUF_BYTES : 0u;
        const unsigned oB = buf ? BBUF_BYTES : 0u;
        #pragma unroll
        for (int j = 0; j < 4; ++j) {
            unsigned long long ga = (unsigned long long)(uintptr_t)(gA[j] + kb);
            asm volatile("global_load_async_to_lds_b128 %0, %1, off"
                         :: "v"(lA[j] + oA), "v"(ga) : "memory");
        }
        #pragma unroll
        for (int j = 0; j < 2; ++j) {
            unsigned long long gb = (unsigned long long)(uintptr_t)(gB[j] + kb);
            asm volatile("global_load_async_to_lds_b128 %0, %1, off"
                         :: "v"(lB[j] + oB), "v"(gb) : "memory");
        }
    };

    v8f acc[4][4] = {};

    auto loadFrag = [&](const unsigned short* base, int row) -> v16bf {
        const unsigned short* p = base + row * LDK + lh * 8;
        union { uint4 q[2]; v16bf v; } u;
        u.q[0] = *(const uint4*)(p);
        u.q[1] = *(const uint4*)(p + 16);
        return u.v;
    };

    auto compute = [&](int buf) {
        const unsigned short* bA = sA + buf * ABUF;
        const unsigned short* bB = sB + buf * BBUF;
        v16bf afr[4], bfr[4];
        #pragma unroll
        for (int mt = 0; mt < 4; ++mt) afr[mt] = loadFrag(bA, wm * 64 + mt * 16 + lr);
        #pragma unroll
        for (int nt = 0; nt < 4; ++nt) bfr[nt] = loadFrag(bB, wn * 64 + nt * 16 + lr);
        #pragma unroll
        for (int mt = 0; mt < 4; ++mt)
            #pragma unroll
            for (int nt = 0; nt < 4; ++nt)
                acc[mt][nt] = __builtin_amdgcn_wmma_f32_16x16x32_bf16(
                    false, afr[mt], false, bfr[nt],
                    (short)0, acc[mt][nt], false, false);
    };

    // ---- software pipeline: 2 stages per trip, buffer parity fixed --------
    issue(0, 0);
    for (int kb = 0; kb < Kdim; kb += 2 * BK) {
        // stage 0: compute buf0, prefetch kb+BK into buf1
        asm volatile("s_wait_asynccnt 0x0" ::: "memory");
        __syncthreads();
        issue(kb + BK, 1);                 // kb+BK <= 4064 < Kdim always
        compute(0);
        // stage 1: compute buf1, prefetch kb+2*BK into buf0
        asm volatile("s_wait_asynccnt 0x0" ::: "memory");
        __syncthreads();
        if (kb + 2 * BK < Kdim) issue(kb + 2 * BK, 0);
        compute(1);
    }

    // ---- epilogue: D layout -> (M = r + 8*lh, N = lr) per 16x16 tile ------
    #pragma unroll
    for (int mt = 0; mt < 4; ++mt) {
        #pragma unroll
        for (int nt = 0; nt < 4; ++nt) {
            const int n     = n0 + wn * 64 + nt * 16 + lr;
            const int mbase = m0 + wm * 64 + mt * 16 + lh * 8;
            const float bv  = bias[n];
            #pragma unroll
            for (int r = 0; r < 8; ++r)
                C[(size_t)(mbase + r) * Ndim + n] = acc[mt][nt][r] + bv;
        }
    }
}

extern "C" void kernel_launch(void* const* d_in, const int* in_sizes, int n_in,
                              void* d_out, int out_size, void* d_ws, size_t ws_size,
                              hipStream_t stream) {
    (void)in_sizes; (void)n_in; (void)out_size; (void)ws_size;
    const float* x    = (const float*)d_in[0];   // [8192,4096]
    const float* w    = (const float*)d_in[1];   // [4096,4096]
    const float* bias = (const float*)d_in[2];   // [4096]
    const float* mask = (const float*)d_in[3];   // [4096,4096]
    float* out = (float*)d_out;                  // [8192,4096]

    unsigned short* xb = (unsigned short*)d_ws;                    // 64 MB
    unsigned short* wb = xb + (size_t)Mdim * Kdim;                 // +32 MB

    pack_x_bf16<<<(Mdim * (size_t)Kdim) / 2048, 256, 0, stream>>>(x, xb);
    pack_w_bf16<<<(Ndim * (size_t)Kdim) / 2048, 256, 0, stream>>>(w, mask, wb);

    dim3 grid(Ndim / 128, Mdim / 256);   // (32, 32)
    sparse_gemm_wmma<<<grid, 256, 0, stream>>>(xb, wb, bias, out);
}
